// BasicMoE_80582176408228
// MI455X (gfx1250) — matrix operand order repaired and checked by reading, not compiled, not run on hardware
//
#include <hip/hip_runtime.h>
#include <hip/hip_bf16.h>
#include <math.h>

// ---------------------------------------------------------------------------
// Dense MoE: out[t,:] = sum_e w[t,e]*(x[t,:] @ W_e) + sum_e w[t,e]*b_e,
//            w = softmax(x @ gate_w + gate_b)
// Pipeline:
//   1) gate kernel -> w[T,8]
//   2) convert x -> bf16 (xbf), convert+transpose expert_w -> bf16 ewT[e][n][k]
//   3) fused GEMM: async-DMA bf16 tiles to LDS (ASYNCcnt, double-buffered),
//      gate-scale A fragments with v_pk_mul_bf16, accumulate all 8 experts
//      into ONE f32 WMMA accumulator (linearity).
// ---------------------------------------------------------------------------

typedef __attribute__((ext_vector_type(16))) __bf16 v16bf;
typedef __attribute__((ext_vector_type(8)))  float  v8f;

#define T_TOK 8192
#define HDIM  1024
#define NEXP  8

#define BM  128
#define BN  128
#define BK  32
#define LSTRIDE 80            // LDS row stride in BYTES (16B aligned, 20-bank skew)
#define LSTRIDE_E 40          // in bf16 elements
#define TILE_BYTES (128 * LSTRIDE)

static __device__ __forceinline__ unsigned f2bf(float f) {
  union { float f; unsigned u; } c; c.f = f;
  unsigned u = c.u + 0x7fffu + ((c.u >> 16) & 1u);   // round-to-nearest-even
  return u >> 16;
}
static __device__ __forceinline__ unsigned pk2f(float a, float b) {
  return f2bf(a) | (f2bf(b) << 16);
}
static __device__ __forceinline__ unsigned pkmul_bf16(unsigned a, unsigned b) {
  unsigned d;
  asm("v_pk_mul_bf16 %0, %1, %2" : "=v"(d) : "v"(a), "v"(b));
  return d;
}
// async 16B copy global -> LDS (tracked by ASYNCcnt)
static __device__ __forceinline__ void async_cp16(unsigned lds_addr, unsigned voff,
                                                  unsigned long long saddr) {
  asm volatile("global_load_async_to_lds_b128 %0, %1, %2"
               :: "v"(lds_addr), "v"(voff), "s"(saddr) : "memory");
}

union Frag { unsigned u[8]; uint4 q[2]; v16bf v; };

// ---------------------------------------------------------------------------
// Kernel 1: gate. One wave per token.
// ---------------------------------------------------------------------------
__global__ __launch_bounds__(256) void moe_gate_kernel(
    const float* __restrict__ x, const float* __restrict__ gw,
    const float* __restrict__ gb, float* __restrict__ wout) {
  const int lane = threadIdx.x & 31;
  const int t    = blockIdx.x * 8 + (threadIdx.x >> 5);
  const float* xr = x + (size_t)t * HDIM;
  float acc[NEXP];
#pragma unroll
  for (int e = 0; e < NEXP; ++e) acc[e] = 0.f;
  for (int h = lane; h < HDIM; h += 32) {
    float xv = xr[h];
    const float4* g4 = (const float4*)(gw + (size_t)h * NEXP);
    float4 g0 = g4[0], g1 = g4[1];
    acc[0] += xv * g0.x; acc[1] += xv * g0.y; acc[2] += xv * g0.z; acc[3] += xv * g0.w;
    acc[4] += xv * g1.x; acc[5] += xv * g1.y; acc[6] += xv * g1.z; acc[7] += xv * g1.w;
  }
#pragma unroll
  for (int off = 16; off > 0; off >>= 1)
#pragma unroll
    for (int e = 0; e < NEXP; ++e) acc[e] += __shfl_xor(acc[e], off, 32);
#pragma unroll
  for (int e = 0; e < NEXP; ++e) acc[e] += gb[e];
  float m = acc[0];
#pragma unroll
  for (int e = 1; e < NEXP; ++e) m = fmaxf(m, acc[e]);
  float s = 0.f;
#pragma unroll
  for (int e = 0; e < NEXP; ++e) { acc[e] = __expf(acc[e] - m); s += acc[e]; }
  float inv = 1.0f / s;
  if (lane == 0) {
    float4* dst = (float4*)(wout + (size_t)t * NEXP);
    dst[0] = make_float4(acc[0]*inv, acc[1]*inv, acc[2]*inv, acc[3]*inv);
    dst[1] = make_float4(acc[4]*inv, acc[5]*inv, acc[6]*inv, acc[7]*inv);
  }
}

// ---------------------------------------------------------------------------
// Kernel 2: x (fp32) -> xbf (bf16), flat. 8 elems/thread.
// ---------------------------------------------------------------------------
__global__ __launch_bounds__(256) void prep_x_kernel(const float* __restrict__ x,
                                                     __bf16* __restrict__ xbf) {
  size_t i8 = ((size_t)blockIdx.x * 256 + threadIdx.x) * 8;
  const float4* s = (const float4*)(x + i8);
  float4 a = s[0], b = s[1];
  uint4 o;
  o.x = pk2f(a.x, a.y); o.y = pk2f(a.z, a.w);
  o.z = pk2f(b.x, b.y); o.w = pk2f(b.z, b.w);
  *(uint4*)(xbf + i8) = o;
}

// ---------------------------------------------------------------------------
// Kernel 3: expert_w [E][k][n] fp32 -> ewT [E][n][k] bf16 (32x32 LDS tiles)
// ---------------------------------------------------------------------------
__global__ __launch_bounds__(256) void prep_wT_kernel(const float* __restrict__ ew,
                                                      __bf16* __restrict__ ewT) {
  __shared__ float t[32][33];
  const int e  = blockIdx.z;
  const int k0 = blockIdx.x * 32;
  const int n0 = blockIdx.y * 32;
  const int r  = threadIdx.x >> 3;
  const int c4 = (threadIdx.x & 7) * 4;
  float4 v = *(const float4*)(ew + ((size_t)e * HDIM + k0 + r) * HDIM + n0 + c4);
  t[r][c4 + 0] = v.x; t[r][c4 + 1] = v.y; t[r][c4 + 2] = v.z; t[r][c4 + 3] = v.w;
  __syncthreads();
  uint2 d;
  d.x = pk2f(t[c4 + 0][r], t[c4 + 1][r]);
  d.y = pk2f(t[c4 + 2][r], t[c4 + 3][r]);
  *(uint2*)(ewT + ((size_t)e * HDIM + n0 + r) * HDIM + k0 + c4) = d;
}

// ---------------------------------------------------------------------------
// Kernel 4: fused expert GEMM + combine.
// Grid (T/BM, H/BN), 256 threads = 8 waves, wave tile 64(M) x 32(N).
// ---------------------------------------------------------------------------
__global__ __launch_bounds__(256) void moe_expert_kernel(
    const __bf16* __restrict__ xbf,  // [T, H]       bf16 row-major
    const __bf16* __restrict__ ewT,  // [E, H(n), H(k)] bf16
    const float*  __restrict__ eb,   // [E, H]
    const float*  __restrict__ w,    // [T, E]
    float* __restrict__ out) {       // [T, H]

  __shared__ __bf16 Alds[128 * LSTRIDE_E];       // 10 KB, single buffer
  __shared__ __bf16 Blds[2 * 128 * LSTRIDE_E];   // 20 KB, double buffer
  __shared__ float  wlds[BM * NEXP];
  __shared__ float  blds[NEXP * BN];

  const int tid   = threadIdx.x;
  const int lane  = tid & 31;
  const int waveM = (tid >> 5) >> 2;
  const int waveN = (tid >> 5) & 3;
  const int m0    = blockIdx.x * BM;
  const int n0    = blockIdx.y * BN;
  const int half  = lane >> 4;
  const int l16   = lane & 15;

  // ---- async copy geometry: 16B chunk id -> (row = id>>2, quad = id&3) ----
  const int crow = tid >> 2;              // 0..63 (chunk 2 adds +64)
  const int cq   = tid & 3;
  const unsigned voffA0 = (unsigned)(((m0 + crow) * HDIM + cq * 8) * 2);
  const unsigned voffB0 = (unsigned)(((n0 + crow) * HDIM + cq * 8) * 2);
  const unsigned voffD  = (unsigned)(64 * HDIM * 2);     // +64 rows
  const unsigned ldsA0  = (unsigned)(size_t)&Alds[0] + crow * LSTRIDE + cq * 16;
  const unsigned ldsB0  = (unsigned)(size_t)&Blds[0] + crow * LSTRIDE + cq * 16;
  const unsigned long long xbase = (unsigned long long)(size_t)xbf;
  const unsigned long long wbase = (unsigned long long)(size_t)ewT;

  // ---- stage gate weights + biases ----
  {
    *(float4*)&wlds[tid * 4] = *(const float4*)(w + (size_t)m0 * NEXP + tid * 4);
    int e  = tid >> 6;
    int j4 = (tid & 63) * 4;
    *(float4*)&blds[e * BN + j4] = *(const float4*)(eb + (size_t)e * HDIM + n0 + j4);
  }

  // ---- prologue: async-stage A(kc=0) and B(e=0,kc=0) into buffer 0 ----
  {
    unsigned long long sa = xbase;               // + kc*2, kc=0
    async_cp16(ldsA0, voffA0, sa);
    async_cp16(ldsA0 + 64 * LSTRIDE, voffA0 + voffD, sa);
    unsigned long long sb = wbase;               // e=0, kc=0
    async_cp16(ldsB0, voffB0, sb);
    async_cp16(ldsB0 + 64 * LSTRIDE, voffB0 + voffD, sb);
  }
  __syncthreads();   // wlds/blds visible

  // per-lane packed gate scalars (row of A-fragment = waveM*64 + mt*16 + l16)
  unsigned wspk[4][NEXP];
#pragma unroll
  for (int mt = 0; mt < 4; ++mt) {
    int rl = waveM * 64 + mt * 16 + l16;
#pragma unroll
    for (int e = 0; e < NEXP; ++e) {
      unsigned b = f2bf(wlds[rl * NEXP + e]);
      wspk[mt][e] = b | (b << 16);
    }
  }

  v8f acc[4][2];
#pragma unroll
  for (int mt = 0; mt < 4; ++mt)
#pragma unroll
    for (int nt = 0; nt < 2; ++nt) acc[mt][nt] = 0.f;

  Frag afr[4];
  const char* AldsB = (const char*)&Alds[0];
  const char* BldsB = (const char*)&Blds[0];

  // ---- main pipeline: i = kc_idx*8 + e, 256 steps ----
#pragma unroll 1
  for (int i = 0; i < 256; ++i) {
    const int e   = i & 7;
    const int buf = i & 1;

    // issue async loads for step i+1 (overlaps with this step's WMMAs)
    const int nxt = i + 1;
    if (nxt < 256) {
      const int kcN = (nxt >> 3) * BK;
      const int eN  = nxt & 7;
      unsigned long long sb =
          wbase + ((unsigned long long)eN * HDIM * HDIM + (unsigned)kcN) * 2ull;
      unsigned bdst = ldsB0 + (nxt & 1) * TILE_BYTES;
      async_cp16(bdst, voffB0, sb);
      async_cp16(bdst + 64 * LSTRIDE, voffB0 + voffD, sb);
      if (eN == 0) {  // new k-chunk: stage next A tile (safe: A frags live in regs)
        unsigned long long sa = xbase + (unsigned)kcN * 2ull;
        async_cp16(ldsA0, voffA0, sa);
        async_cp16(ldsA0 + 64 * LSTRIDE, voffA0 + voffD, sa);
        asm volatile("s_wait_asynccnt 4" ::: "memory");
      } else {
        asm volatile("s_wait_asynccnt 2" ::: "memory");
      }
    } else {
      asm volatile("s_wait_asynccnt 0" ::: "memory");
    }
    __syncthreads();   // current-step tiles visible to all waves

    if (e == 0) {      // build A fragments once per k-chunk; reuse for 8 experts
#pragma unroll
      for (int mt = 0; mt < 4; ++mt) {
        const char* base = AldsB + (waveM * 64 + mt * 16 + l16) * LSTRIDE;
        afr[mt].q[0] = *(const uint4*)(base + 16 * half);        // K 0..7  (+8*half)
        afr[mt].q[1] = *(const uint4*)(base + 32 + 16 * half);   // K 16..23(+8*half)
      }
    }

    Frag bfr[2];
#pragma unroll
    for (int nt = 0; nt < 2; ++nt) {
      const char* base = BldsB + buf * TILE_BYTES +
                         (waveN * 32 + nt * 16 + l16) * LSTRIDE + 16 * half;
      bfr[nt].q[0] = *(const uint4*)(base);        // K 0..7  (+8*half)
      bfr[nt].q[1] = *(const uint4*)(base + 32);   // K 16..23(+8*half)
    }

#pragma unroll
    for (int mt = 0; mt < 4; ++mt) {
      Frag as;  // gate-scaled A fragment: one scalar per lane (its row)
#pragma unroll
      for (int v = 0; v < 8; ++v) as.u[v] = pkmul_bf16(afr[mt].u[v], wspk[mt][e]);
#pragma unroll
      for (int nt = 0; nt < 2; ++nt) {
        acc[mt][nt] = __builtin_amdgcn_wmma_f32_16x16x32_bf16(
            false, as.v, false, bfr[nt].v, (short)0, acc[mt][nt], false, false);
      }
    }
    __syncthreads();   // compute done before buffers are overwritten
  }

  // ---- epilogue: add sum_e w[t,e]*b_e[n], store ----
#pragma unroll
  for (int mt = 0; mt < 4; ++mt) {
#pragma unroll
    for (int i = 0; i < 8; ++i) {
      int rl = waveM * 64 + mt * 16 + i + 8 * half;
      float wr[NEXP];
#pragma unroll
      for (int e = 0; e < NEXP; ++e) wr[e] = wlds[rl * NEXP + e];
#pragma unroll
      for (int nt = 0; nt < 2; ++nt) {
        int cl = waveN * 32 + nt * 16 + l16;
        float bias = 0.f;
#pragma unroll
        for (int e = 0; e < NEXP; ++e) bias += wr[e] * blds[e * BN + cl];
        out[(size_t)(m0 + rl) * HDIM + n0 + cl] = acc[mt][nt][i] + bias;
      }
    }
  }
}

// ---------------------------------------------------------------------------
extern "C" void kernel_launch(void* const* d_in, const int* in_sizes, int n_in,
                              void* d_out, int out_size, void* d_ws, size_t ws_size,
                              hipStream_t stream) {
  const float* x  = (const float*)d_in[0];   // [B,S,H] fp32
  const float* gw = (const float*)d_in[1];   // [H,E]
  const float* gb = (const float*)d_in[2];   // [E]
  const float* ew = (const float*)d_in[3];   // [E,H,H]
  const float* eb = (const float*)d_in[4];   // [E,H]
  float* out = (float*)d_out;

  // workspace layout: w[T,8] f32 | xbf[T,H] bf16 | ewT[E,H,H] bf16
  char* wsb = (char*)d_ws;
  float*  wbuf = (float*)wsb;
  __bf16* xbf  = (__bf16*)(wsb + ((size_t)T_TOK * NEXP * 4 + 255 & ~(size_t)255));
  __bf16* ewT  = xbf + (size_t)T_TOK * HDIM;

  moe_gate_kernel<<<T_TOK / 8, 256, 0, stream>>>(x, gw, gb, wbuf);
  prep_x_kernel<<<(T_TOK * HDIM) / (256 * 8), 256, 0, stream>>>(x, xbf);
  prep_wT_kernel<<<dim3(HDIM / 32, HDIM / 32, NEXP), 256, 0, stream>>>(ew, ewT);
  moe_expert_kernel<<<dim3(T_TOK / BM, HDIM / BN), 256, 0, stream>>>(xbf, ewT, eb, wbuf, out);
}
